// GATModelPaper_893353198376
// MI455X (gfx1250) — compile-verified, matrix operand order checked by reading, not run on hardware
//
#include <hip/hip_runtime.h>

#define NN   50000
#define NE   1600000
#define ETOT (NE + NN)     // edges + self loops
#define INC  512
#define C1   64            // HEADS1 * HID1
#define OUTC 40

typedef __attribute__((ext_vector_type(16))) _Float16 v16h;
typedef __attribute__((ext_vector_type(8)))  float    v8f;

// ---- order-preserving float <-> uint map for atomicMax on floats ----
static __device__ __forceinline__ unsigned f2ord(float x) {
  unsigned u = __float_as_uint(x);
  return (u & 0x80000000u) ? ~u : (u | 0x80000000u);
}
static __device__ __forceinline__ float ord2f(unsigned u) {
  return (u & 0x80000000u) ? __uint_as_float(u & 0x7fffffffu)
                           : __uint_as_float(~u);
}

// ---- zero scratch (m/den/acc regions) ----
__global__ void k_zero(unsigned* __restrict__ p, int n) {
  int i = blockIdx.x * blockDim.x + threadIdx.x;
  for (; i < n; i += gridDim.x * blockDim.x) p[i] = 0u;
}

// ---- row-sum reciprocal: one wave32 per row ----
__global__ void __launch_bounds__(256) k_rownorm(const float* __restrict__ x,
                                                 float* __restrict__ inv_rs) {
  int wave = threadIdx.x >> 5, lane = threadIdx.x & 31;
  int row = blockIdx.x * 8 + wave;
  if (row >= NN) return;
  const float* xr = x + (size_t)row * INC;
  float s = 0.f;
#pragma unroll
  for (int i = 0; i < INC / 32; ++i) s += xr[lane + i * 32];
#pragma unroll
  for (int off = 16; off; off >>= 1) s += __shfl_down(s, off, 32);
  if (lane == 0) inv_rs[row] = 1.f / fmaxf(s, 1e-8f);
}

// ---- GEMM1: h1[N,64] = (x * inv_rs) @ W1, f16 WMMA, f32 accum ----
__global__ void __launch_bounds__(256) k_gemm1(const float* __restrict__ x,
                                               const float* __restrict__ inv_rs,
                                               const float* __restrict__ W1,
                                               float* __restrict__ h1) {
  __shared__ _Float16 bs[16 * 4 * 512 / 16 * 16]; // 32768 halves = 64KB, frag-order
  int tid = threadIdx.x;
  // stage W1 -> LDS pre-swizzled: [tile(ks*4+nt)][lane][j]
  for (int idx = tid; idx < 32768; idx += 256) {
    int j = idx & 15;
    int L = (idx >> 4) & 31;
    int tile = idx >> 9;                 // 0..63
    int ks = tile >> 2, nt = tile & 3;
    int k = ks * 32 + ((j >> 3) << 4) + ((L >> 4) << 3) + (j & 7);
    int n = nt * 16 + (L & 15);
    bs[idx] = (_Float16)W1[k * 64 + n];
  }
  __syncthreads();
  int wave = tid >> 5, lane = tid & 31;
  int l16 = lane & 15, hi = lane >> 4;
  int mbase = (blockIdx.x * 8 + wave) * 16;
  int row = mbase + l16;
  int rowc = row < NN ? row : 0;
  float scale = row < NN ? inv_rs[rowc] : 0.f;
  const float* ar = x + (size_t)rowc * INC + hi * 8;
  const v16h* bv = (const v16h*)bs;
  v8f acc0 = {}, acc1 = {}, acc2 = {}, acc3 = {};
  for (int ks = 0; ks < 16; ++ks) {
    int k0 = ks * 32;
    v16h a;
#pragma unroll
    for (int j = 0; j < 8; ++j) a[j] = (_Float16)(ar[k0 + j] * scale);
#pragma unroll
    for (int j = 0; j < 8; ++j) a[8 + j] = (_Float16)(ar[k0 + 16 + j] * scale);
    v16h b0 = bv[(ks * 4 + 0) * 32 + lane];
    v16h b1 = bv[(ks * 4 + 1) * 32 + lane];
    v16h b2 = bv[(ks * 4 + 2) * 32 + lane];
    v16h b3 = bv[(ks * 4 + 3) * 32 + lane];
    acc0 = __builtin_amdgcn_wmma_f32_16x16x32_f16(false, a, false, b0, (short)0, acc0, false, false);
    acc1 = __builtin_amdgcn_wmma_f32_16x16x32_f16(false, a, false, b1, (short)0, acc1, false, false);
    acc2 = __builtin_amdgcn_wmma_f32_16x16x32_f16(false, a, false, b2, (short)0, acc2, false, false);
    acc3 = __builtin_amdgcn_wmma_f32_16x16x32_f16(false, a, false, b3, (short)0, acc3, false, false);
  }
#pragma unroll
  for (int r = 0; r < 8; ++r) {
    int orow = mbase + hi * 8 + r;
    if (orow < NN) {
      float* o = h1 + (size_t)orow * 64 + l16;
      o[0] = acc0[r]; o[16] = acc1[r]; o[32] = acc2[r]; o[48] = acc3[r];
    }
  }
}

// ---- GEMM2: h2[N,40] = h1_out @ W2, f16 WMMA ----
__global__ void __launch_bounds__(256) k_gemm2(const float* __restrict__ h1o,
                                               const float* __restrict__ W2,
                                               float* __restrict__ h2) {
  __shared__ _Float16 bs[6 * 512]; // 2 ksteps * 3 ntiles, frag-order
  int tid = threadIdx.x;
  for (int idx = tid; idx < 3072; idx += 256) {
    int j = idx & 15;
    int L = (idx >> 4) & 31;
    int tile = idx >> 9;                 // 0..5
    int ks = tile / 3, nt = tile % 3;
    int k = ks * 32 + ((j >> 3) << 4) + ((L >> 4) << 3) + (j & 7);
    int n = nt * 16 + (L & 15);
    bs[idx] = (n < OUTC) ? (_Float16)W2[k * OUTC + n] : (_Float16)0.f;
  }
  __syncthreads();
  int wave = tid >> 5, lane = tid & 31;
  int l16 = lane & 15, hi = lane >> 4;
  int mbase = (blockIdx.x * 8 + wave) * 16;
  int row = mbase + l16;
  int rowc = row < NN ? row : 0;
  float scale = row < NN ? 1.f : 0.f;
  const float* ar = h1o + (size_t)rowc * 64 + hi * 8;
  const v16h* bv = (const v16h*)bs;
  v8f acc0 = {}, acc1 = {}, acc2 = {};
#pragma unroll
  for (int ks = 0; ks < 2; ++ks) {
    int k0 = ks * 32;
    v16h a;
#pragma unroll
    for (int j = 0; j < 8; ++j) a[j] = (_Float16)(ar[k0 + j] * scale);
#pragma unroll
    for (int j = 0; j < 8; ++j) a[8 + j] = (_Float16)(ar[k0 + 16 + j] * scale);
    v16h b0 = bv[(ks * 3 + 0) * 32 + lane];
    v16h b1 = bv[(ks * 3 + 1) * 32 + lane];
    v16h b2 = bv[(ks * 3 + 2) * 32 + lane];
    acc0 = __builtin_amdgcn_wmma_f32_16x16x32_f16(false, a, false, b0, (short)0, acc0, false, false);
    acc1 = __builtin_amdgcn_wmma_f32_16x16x32_f16(false, a, false, b1, (short)0, acc1, false, false);
    acc2 = __builtin_amdgcn_wmma_f32_16x16x32_f16(false, a, false, b2, (short)0, acc2, false, false);
  }
#pragma unroll
  for (int r = 0; r < 8; ++r) {
    int orow = mbase + hi * 8 + r;
    if (orow < NN) {
      float v[3] = {acc0[r], acc1[r], acc2[r]};
#pragma unroll
      for (int nt = 0; nt < 3; ++nt) {
        int col = nt * 16 + l16;
        if (col < OUTC) h2[(size_t)orow * OUTC + col] = v[nt];
      }
    }
  }
}

// ---- attention logits per node ----
__global__ void k_att1(const float* __restrict__ h1, const float* __restrict__ asw,
                       const float* __restrict__ adw, float* __restrict__ a_src,
                       float* __restrict__ a_dst) {
  int i = blockIdx.x * blockDim.x + threadIdx.x;  // node*8 + head
  if (i >= NN * 8) return;
  int n = i >> 3, h = i & 7;
  const float* hp = h1 + (size_t)n * 64 + h * 8;
  float s = 0.f, t = 0.f;
#pragma unroll
  for (int c = 0; c < 8; ++c) { float v = hp[c]; s += v * asw[h * 8 + c]; t += v * adw[h * 8 + c]; }
  a_src[i] = s; a_dst[i] = t;
}

__global__ void k_att2(const float* __restrict__ h2, const float* __restrict__ asw,
                       const float* __restrict__ adw, float* __restrict__ a_src,
                       float* __restrict__ a_dst) {
  int n = blockIdx.x * blockDim.x + threadIdx.x;
  if (n >= NN) return;
  const float* hp = h2 + (size_t)n * OUTC;
  float s = 0.f, t = 0.f;
#pragma unroll
  for (int c = 0; c < OUTC; ++c) { float v = hp[c]; s += v * asw[c]; t += v * adw[c]; }
  a_src[n] = s; a_dst[n] = t;
}

// ---- edge pass 1: segment max of leaky_relu logits ----
template <int H>
__global__ void __launch_bounds__(256) k_edge_max(const long long* __restrict__ ei,
                                                  const float* __restrict__ a_src,
                                                  const float* __restrict__ a_dst,
                                                  unsigned* __restrict__ m) {
  int i = blockIdx.x * blockDim.x + threadIdx.x;
  if (i >= ETOT * H) return;
  int e = i / H, h = i % H;
  int s, d;
  if (e < NE) { s = (int)ei[e]; d = (int)ei[NE + e]; } else { s = e - NE; d = s; }
  float al = a_src[s * H + h] + a_dst[d * H + h];
  al = al > 0.f ? al : 0.2f * al;
  atomicMax(&m[d * H + h], f2ord(al));
}

// ---- edge pass 2: exp weights -> denom + weighted feature accumulation ----
template <int H, int C>
__global__ void __launch_bounds__(256) k_edge_sum(const long long* __restrict__ ei,
                                                  const float* __restrict__ a_src,
                                                  const float* __restrict__ a_dst,
                                                  const unsigned* __restrict__ m,
                                                  const float* __restrict__ hsrc,
                                                  float* __restrict__ den,
                                                  float* __restrict__ acc) {
  int i = blockIdx.x * blockDim.x + threadIdx.x;
  if (i >= ETOT * H) return;
  int e = i / H, h = i % H;
  int s, d;
  if (e < NE) { s = (int)ei[e]; d = (int)ei[NE + e]; } else { s = e - NE; d = s; }
  float al = a_src[s * H + h] + a_dst[d * H + h];
  al = al > 0.f ? al : 0.2f * al;
  float w = __expf(al - ord2f(m[d * H + h]));
  atomicAdd(&den[d * H + h], w);
  const float* hp = hsrc + (size_t)s * (H * C) + h * C;
  float* ap = acc + (size_t)d * (H * C) + h * C;
#pragma unroll
  for (int c = 0; c < C; ++c) atomicAdd(ap + c, w * hp[c]);
}

// ---- finalize: divide by denom, add bias, optional ELU ----
template <int H, int C, bool ELU>
__global__ void k_fin(const float* __restrict__ acc, const float* __restrict__ den,
                      const float* __restrict__ bias, float* __restrict__ out) {
  int i = blockIdx.x * blockDim.x + threadIdx.x;
  if (i >= NN * H * C) return;
  int n = i / (H * C);
  int r = i - n * (H * C);
  int h = r / C;
  float v = acc[i] / (den[n * H + h] + 1e-16f) + bias[r];
  if (ELU) v = v > 0.f ? v : (__expf(v) - 1.f);
  out[i] = v;
}

extern "C" void kernel_launch(void* const* d_in, const int* in_sizes, int n_in,
                              void* d_out, int out_size, void* d_ws, size_t ws_size,
                              hipStream_t stream) {
  const float*     x    = (const float*)d_in[0];
  const long long* ei   = (const long long*)d_in[1];  // int64 per reference
  const float*     W1   = (const float*)d_in[2];
  const float*     as1w = (const float*)d_in[3];
  const float*     ad1w = (const float*)d_in[4];
  const float*     b1   = (const float*)d_in[5];
  const float*     W2   = (const float*)d_in[6];
  const float*     as2w = (const float*)d_in[7];
  const float*     ad2w = (const float*)d_in[8];
  const float*     b2   = (const float*)d_in[9];
  float*           out  = (float*)d_out;

  // workspace layout (f32 words)
  float* w = (float*)d_ws;
  float*    inv_rs = w;                                  // N
  float*    h1     = inv_rs + NN;                        // N*64
  float*    as1    = h1 + (size_t)NN * 64;               // N*8
  float*    ad1    = as1 + (size_t)NN * 8;               // N*8
  float*    h2     = ad1 + (size_t)NN * 8;               // N*40
  float*    as2    = h2 + (size_t)NN * OUTC;             // N
  float*    ad2    = as2 + NN;                           // N
  unsigned* m1     = (unsigned*)(ad2 + NN);              // N*8   (zeroed)
  float*    den1   = (float*)(m1 + (size_t)NN * 8);      // N*8   (zeroed)
  float*    acc1   = den1 + (size_t)NN * 8;              // N*64  (zeroed; becomes h1_out)
  unsigned* m2     = (unsigned*)(acc1 + (size_t)NN * 64);// N     (zeroed)
  float*    den2   = (float*)(m2 + NN);                  // N     (zeroed)
  float*    acc2   = den2 + NN;                          // N*40  (zeroed)
  int nzero = NN * (8 + 8 + 64 + 1 + 1 + OUTC);

  k_zero<<<2048, 256, 0, stream>>>(m1, nzero);
  k_rownorm<<<(NN + 7) / 8, 256, 0, stream>>>(x, inv_rs);

  // layer 1
  k_gemm1<<<(NN + 127) / 128, 256, 0, stream>>>(x, inv_rs, W1, h1);
  k_att1<<<(NN * 8 + 255) / 256, 256, 0, stream>>>(h1, as1w, ad1w, as1, ad1);
  k_edge_max<8><<<(ETOT * 8 + 255) / 256, 256, 0, stream>>>(ei, as1, ad1, m1);
  k_edge_sum<8, 8><<<(ETOT * 8 + 255) / 256, 256, 0, stream>>>(ei, as1, ad1, m1, h1, den1, acc1);
  k_fin<8, 8, true><<<(NN * 64 + 255) / 256, 256, 0, stream>>>(acc1, den1, b1, acc1);

  // layer 2
  k_gemm2<<<(NN + 127) / 128, 256, 0, stream>>>(acc1, W2, h2);
  k_att2<<<(NN + 255) / 256, 256, 0, stream>>>(h2, as2w, ad2w, as2, ad2);
  k_edge_max<1><<<(ETOT + 255) / 256, 256, 0, stream>>>(ei, as2, ad2, m2);
  k_edge_sum<1, OUTC><<<(ETOT + 255) / 256, 256, 0, stream>>>(ei, as2, ad2, m2, h2, den2, acc2);
  k_fin<1, OUTC, false><<<(NN * OUTC + 255) / 256, 256, 0, stream>>>(acc2, den2, b2, out);
}
